// PreCoBlock_18459769438648
// MI455X (gfx1250) — compile-verified, matrix operand order checked by reading, not run on hardware
//
#include <hip/hip_runtime.h>
#include <math.h>

// ---------------- problem constants (from the reference) ----------------
constexpr int B_  = 4;
constexpr int L_  = 2048;
constexpr int DM  = 512;                 // D_MODEL
constexpr int DI  = 3072;                // D_INNER
constexpr int DTR = 32;                  // DT_RANK
constexpr int DST = 8;                   // D_STATE
constexpr int H_  = 8;                   // N_HEADS
constexpr int HD  = 64;                  // HEAD_DIM
constexpr int MBS = 8;                   // MB
constexpr int NMB = L_ / MBS;            // 256
constexpr int WINR = 5;
constexpr int XZN = 2 * DI;              // 6144
constexpr int XDN = DTR + 2 * DST;       // 48

typedef __attribute__((ext_vector_type(16))) __bf16 v16bf;
typedef __attribute__((ext_vector_type(4)))  __bf16 v4bf;
typedef __attribute__((ext_vector_type(8)))  float  v8f;

__device__ __forceinline__ float sigm(float x) { return 1.f / (1.f + expf(-x)); }

// ======================================================================
// Generic row LayerNorm: out[r,:] = (x - mu) * rsqrt(var+eps) * w + b
// ======================================================================
__global__ __launch_bounds__(256) void ln_kernel(const float* __restrict__ in,
                                                 const float* __restrict__ w,
                                                 const float* __restrict__ bias,
                                                 float* __restrict__ out, int width) {
  const int row = blockIdx.x;
  const float* p = in + (size_t)row * width;
  __shared__ float s1[256], s2[256];
  float a = 0.f, b = 0.f;
  for (int i = threadIdx.x; i < width; i += blockDim.x) {
    float v = p[i]; a += v; b += v * v;
  }
  s1[threadIdx.x] = a; s2[threadIdx.x] = b;
  __syncthreads();
  for (int st = 128; st > 0; st >>= 1) {
    if ((int)threadIdx.x < st) { s1[threadIdx.x] += s1[threadIdx.x + st]; s2[threadIdx.x] += s2[threadIdx.x + st]; }
    __syncthreads();
  }
  const float mu = s1[0] / width;
  const float var = s2[0] / width - mu * mu;
  const float rstd = rsqrtf(var + 1e-6f);
  float* o = out + (size_t)row * width;
  for (int i = threadIdx.x; i < width; i += blockDim.x)
    o[i] = (p[i] - mu) * rstd * w[i] + bias[i];
}

// ======================================================================
// bf16 WMMA GEMM:  C[M,N] = A[M,K] * W[N,K]^T     (f32 in/out, bf16 MACs)
// Requirements: M % 128 == 0, K % 32 == 0 (true for every call site).
// Block tile 128x64, 8 waves (4x2), each wave 32x32 via 2x2 WMMA frags.
// float4 global loads staged through registers and software-pipelined so
// the next tile's loads overlap the current tile's v_wmma issue.
// A-fragment layout per CDNA5 ISA (16-bit A 16x32): lane%16 = M row,
// element e -> K via {d=e/2,p=e&1; k=(d<4?2d+p:16+2(d-4)+p)+8*(lane/16)}.
// B assumed symmetric: lane%16 = N column, same K-per-element pattern.
// C/D: N = lane%16, M = vgpr + 8*(lane/16).
// ======================================================================
#define BM 128
#define BN 64
#define BK 32
#define LDSP 40   // padded LDS row stride (bf16 elems); col*4 stays b64-aligned

__global__ __launch_bounds__(256) void gemm_bf16_kernel(const float* __restrict__ A,
                                                        const float* __restrict__ W,
                                                        float* __restrict__ C,
                                                        int M, int N, int K, int lda) {
  __shared__ __bf16 As[BM * LDSP];
  __shared__ __bf16 Bs[BN * LDSP];
  const int tid  = threadIdx.x;
  const int lane = tid & 31;
  const int wave = tid >> 5;
  const int waveM = wave & 3;     // 0..3
  const int waveN = wave >> 2;    // 0..1
  const int bM = blockIdx.y * BM;
  const int bN = blockIdx.x * BN;

  v8f acc[2][2];
#pragma unroll
  for (int i = 0; i < 2; i++)
#pragma unroll
    for (int j = 0; j < 2; j++)
#pragma unroll
      for (int e = 0; e < 8; e++) acc[i][j][e] = 0.f;

  int kk[16];
#pragma unroll
  for (int e = 0; e < 16; e++) {
    int d = e >> 1, p = e & 1;
    int base = (d < 4) ? (2 * d + p) : (16 + 2 * (d - 4) + p);
    kk[e] = base + ((lane >= 16) ? 8 : 0);
  }
  const int lrow = lane & 15;

  // register staging for the software pipeline
  float4 aReg[4];   // A tile: 128x32 f32 = 1024 float4, 4 per thread
  float4 bReg[2];   // B tile:  64x32 f32 =  512 float4, 2 per thread

  auto loadTiles = [&](int k0) {
#pragma unroll
    for (int i = 0; i < 4; i++) {
      const int idx = i * 256 + tid;            // 0..1023
      const int r = idx >> 3, cv = idx & 7;     // row, float4-col
      aReg[i] = *(const float4*)(A + (size_t)(bM + r) * lda + k0 + cv * 4);
    }
#pragma unroll
    for (int i = 0; i < 2; i++) {
      const int idx = i * 256 + tid;            // 0..511
      const int r = idx >> 3, cv = idx & 7;
      const int gn = bN + r;
      const int gc = gn < N ? gn : N - 1;       // clamp (branchless OOB)
      float4 v = *(const float4*)(W + (size_t)gc * K + k0 + cv * 4);
      if (gn >= N) { v.x = 0.f; v.y = 0.f; v.z = 0.f; v.w = 0.f; }
      bReg[i] = v;
    }
  };
  auto stageTiles = [&]() {
#pragma unroll
    for (int i = 0; i < 4; i++) {
      const int idx = i * 256 + tid;
      const int r = idx >> 3, cv = idx & 7;
      v4bf pk;
      pk[0] = (__bf16)aReg[i].x; pk[1] = (__bf16)aReg[i].y;
      pk[2] = (__bf16)aReg[i].z; pk[3] = (__bf16)aReg[i].w;
      *(v4bf*)&As[r * LDSP + cv * 4] = pk;
    }
#pragma unroll
    for (int i = 0; i < 2; i++) {
      const int idx = i * 256 + tid;
      const int r = idx >> 3, cv = idx & 7;
      v4bf pk;
      pk[0] = (__bf16)bReg[i].x; pk[1] = (__bf16)bReg[i].y;
      pk[2] = (__bf16)bReg[i].z; pk[3] = (__bf16)bReg[i].w;
      *(v4bf*)&Bs[r * LDSP + cv * 4] = pk;
    }
  };

  loadTiles(0);
  for (int k0 = 0; k0 < K; k0 += BK) {
    stageTiles();
    __syncthreads();
    if (k0 + BK < K) loadTiles(k0 + BK);   // overlap next loads with WMMA

    v16bf aF[2], bF[2];
#pragma unroll
    for (int mt = 0; mt < 2; mt++) {
      const int m = waveM * 32 + mt * 16 + lrow;
#pragma unroll
      for (int e = 0; e < 16; e++) aF[mt][e] = As[m * LDSP + kk[e]];
    }
#pragma unroll
    for (int nt = 0; nt < 2; nt++) {
      const int n = waveN * 32 + nt * 16 + lrow;
#pragma unroll
      for (int e = 0; e < 16; e++) bF[nt][e] = Bs[n * LDSP + kk[e]];
    }
#pragma unroll
    for (int mt = 0; mt < 2; mt++)
#pragma unroll
      for (int nt = 0; nt < 2; nt++)
        acc[mt][nt] = __builtin_amdgcn_wmma_f32_16x16x32_bf16(
            false, aF[mt], false, bF[nt], (short)0, acc[mt][nt], false, false);
    __syncthreads();
  }

  const int mHalf = (lane >= 16) ? 8 : 0;
  const int nCol  = lane & 15;
#pragma unroll
  for (int mt = 0; mt < 2; mt++)
#pragma unroll
    for (int nt = 0; nt < 2; nt++) {
      const int gn = bN + waveN * 32 + nt * 16 + nCol;
      if (gn >= N) continue;
#pragma unroll
      for (int v = 0; v < 8; v++) {
        const int gm = bM + waveM * 32 + mt * 16 + mHalf + v;
        C[(size_t)gm * N + gn] = acc[mt][nt][v];
      }
    }
}

// ======================================================================
// Depthwise conv (width 3, causal pad 2) + SiLU:  xz[:, :, :DI] -> xc
// ======================================================================
__global__ void conv_silu_kernel(const float* __restrict__ xz,
                                 const float* __restrict__ cw,
                                 const float* __restrict__ cb,
                                 float* __restrict__ xc) {
  int idx = blockIdx.x * blockDim.x + threadIdx.x;
  if (idx >= B_ * L_ * DI) return;
  const int c = idx % DI;
  int t = idx / DI;
  const int l = t % L_;
  const int b = t / L_;
  float acc = cb[c];
#pragma unroll
  for (int w = 0; w < 3; w++) {
    const int tt = l + w - 2;
    if (tt >= 0) acc += xz[((size_t)(b * L_ + tt)) * XZN + c] * cw[c * 3 + w];
  }
  xc[idx] = acc * sigm(acc);
}

// ======================================================================
// Longhorn scan, sequential over L; thread = one (b, channel); 8-state
// in registers. Operands for step t+1 are prefetched while computing t.
// Fuses the silu(z) output gating. y written over xc.
// ======================================================================
__global__ __launch_bounds__(256) void scan_kernel(float* __restrict__ xc,
                                                   const float* __restrict__ dtv,
                                                   const float* __restrict__ xdbl,
                                                   const float* __restrict__ xz,
                                                   const float* __restrict__ dtb,
                                                   const float* __restrict__ Dp) {
  const int c = blockIdx.x * blockDim.x + threadIdx.x;
  const int b = blockIdx.y;
  float S[DST];
#pragma unroll
  for (int n = 0; n < DST; n++) S[n] = 0.f;
  const float dtbc = dtb[c];
  const float Dc = Dp[c];
  __shared__ float kq[16];

  // prefetch t = 0
  size_t r0 = (size_t)b * L_;
  float x_n  = xc[r0 * DI + c];
  float dv_n = dtv[r0 * DI + c];
  float zv_n = xz[r0 * XZN + DI + c];
  float kq_n = 0.f;
  if (threadIdx.x < 16) kq_n = xdbl[r0 * XDN + DTR + threadIdx.x];

  for (int t = 0; t < L_; t++) {
    const size_t row = (size_t)(b * L_ + t);
    const float x = x_n, dv = dv_n, zv = zv_n;
    if (threadIdx.x < 16) kq[threadIdx.x] = kq_n;
    __syncthreads();

    if (t + 1 < L_) {              // prefetch t+1 (overlaps compute below)
      const size_t r2 = row + 1;
      x_n  = xc[r2 * DI + c];
      dv_n = dtv[r2 * DI + c];
      zv_n = xz[r2 * XZN + DI + c];
      if (threadIdx.x < 16) kq_n = xdbl[r2 * XDN + DTR + threadIdx.x];
    }

    float kk[DST], qq[DST], sk2 = 0.f;
#pragma unroll
    for (int n = 0; n < DST; n++) { kk[n] = kq[n]; qq[n] = kq[DST + n]; sk2 += kk[n] * kk[n]; }
    float d = sigm(dv + dtbc);
    d = d / (1.f + d * sk2);
    float y = 0.f;
#pragma unroll
    for (int n = 0; n < DST; n++) {
      const float k2 = kk[n] * kk[n];
      S[n] = S[n] * (1.f - d * k2) + (d * x) * kk[n];
      y += S[n] * qq[n];
    }
    y += Dc * x;
    xc[row * DI + c] = y * (zv * sigm(zv));
    __syncthreads();
  }
}

// ======================================================================
// In-place rotary on a (B,L,H*HD) tensor, per-head dim 64
// ======================================================================
__global__ void rotary_kernel(float* __restrict__ X) {
  int idx = blockIdx.x * blockDim.x + threadIdx.x;
  if (idx >= B_ * L_ * H_ * (HD / 2)) return;
  const int d = idx % 32;
  int t = idx / 32;
  const int h = t % H_; t /= H_;
  const int l = t % L_;
  const int b = t / L_;
  const float inv = powf(10000.f, -(float)d / 32.f);
  const float ang = (float)l * inv;
  const float cs = cosf(ang), sn = sinf(ang);
  const size_t base = ((size_t)(b * L_ + l)) * DM + h * HD;
  const float x1 = X[base + d], x2 = X[base + d + 32];
  X[base + d]      = x1 * cs - x2 * sn;
  X[base + d + 32] = x2 * cs + x1 * sn;
}

// ======================================================================
// lr projection: lrv[b,h,l] = sigmoid(xn[b,l,:] . lr_w[h,0,:] + lr_b[h])
// (8 consecutive threads share one xn row -> L2/L0 friendly)
// ======================================================================
__global__ void lr_kernel(const float* __restrict__ xn, const float* __restrict__ lrw,
                          const float* __restrict__ lrb, float* __restrict__ lrv) {
  int idx = blockIdx.x * blockDim.x + threadIdx.x;
  if (idx >= B_ * L_ * H_) return;
  const int h = idx % H_;
  int t = idx / H_;
  const int l = t % L_;
  const int b = t / L_;
  const float* xr = xn + ((size_t)(b * L_ + l)) * DM;
  const float* wr = lrw + (size_t)h * DM;
  float s = 0.f;
  for (int c = 0; c < DM; c++) s += xr[c] * wr[c];
  lrv[((size_t)(b * H_ + h)) * L_ + l] = sigm(s + lrb[h]);
}

// ======================================================================
// TTT-linear scan: one block per (b,h); 64 threads, thread j owns column
// j of the 64x64 fast-weight W1 (kept in LDS; column is thread-private).
// Sequential over 256 minibatches of 8 tokens.
// ======================================================================
__global__ __launch_bounds__(64) void ttt_scan_kernel(const float* __restrict__ XQ,
                                                      const float* __restrict__ XK,
                                                      const float* __restrict__ XV,
                                                      const float* __restrict__ lrv,
                                                      const float* __restrict__ tdelta,
                                                      const float* __restrict__ W1p,
                                                      const float* __restrict__ b1p,
                                                      const float* __restrict__ nw,
                                                      const float* __restrict__ nb,
                                                      float* __restrict__ outp) {
  const int bh = blockIdx.x;
  const int b = bh / H_, h = bh % H_;
  const int j = threadIdx.x;
  __shared__ float W1s[HD][HD + 1];
  __shared__ float xq[MBS][HD], xk[MBS][HD], xv[MBS][HD];
  __shared__ float zs[MBS][HD], gxh[MBS][HD], xh[MBS][HD];
  __shared__ float mu[MBS], sd[MBS], m1[MBS], m2[MBS];
  __shared__ float attn[MBS][MBS], co[MBS][MBS], lr8[MBS], tf[MBS];

  for (int d = 0; d < HD; d++) W1s[d][j] = W1p[((size_t)h * HD + d) * HD + j];
  float b1j = b1p[h * HD + j];
  const float gj = nw[h * HD + j], bj = nb[h * HD + j];
  if (j < MBS) tf[j] = fmaxf(1.f / (float)(j + 1) + tdelta[j], 0.f);
  __syncthreads();

  for (int mb = 0; mb < NMB; mb++) {
    const int l0 = mb * MBS;
    for (int i = 0; i < MBS; i++) {
      const size_t base = ((size_t)(b * L_ + l0 + i)) * DM + h * HD + j;
      xq[i][j] = XQ[base]; xk[i][j] = XK[base]; xv[i][j] = XV[base];
    }
    if (j < MBS) lr8[j] = lrv[((size_t)(b * H_ + h)) * L_ + l0 + j];
    __syncthreads();

    { // attn = tril(xq @ xk^T), one thread per (i,k)
      const int i = j >> 3, k = j & 7;
      float s = 0.f;
      if (k <= i) for (int d = 0; d < HD; d++) s += xq[i][d] * xk[k][d];
      attn[i][k] = (k <= i) ? s : 0.f;
    }
    float z[MBS];
    for (int i = 0; i < MBS; i++) {          // Z1 column j
      float s = b1j;
      for (int d = 0; d < HD; d++) s += xk[i][d] * W1s[d][j];
      z[i] = s; zs[i][j] = s;
    }
    __syncthreads();
    if (j < MBS) {                           // Z1 row stats
      float s = 0.f, q = 0.f;
      for (int d = 0; d < HD; d++) { const float v = zs[j][d]; s += v; q += v * v; }
      const float m = s / HD; mu[j] = m; sd[j] = sqrtf(q / HD - m * m + 1e-6f);
    }
    __syncthreads();
    for (int i = 0; i < MBS; i++) {          // ln_l2_bwd part 1
      const float xhat = (z[i] - mu[i]) / sd[i];
      xh[i][j] = xhat;
      const float tgt = xv[i][j] - xk[i][j];
      gxh[i][j] = (gj * xhat + bj - tgt) * gj;
    }
    __syncthreads();
    if (j < MBS) {
      float s = 0.f, q = 0.f;
      for (int d = 0; d < HD; d++) { s += gxh[j][d]; q += gxh[j][d] * xh[j][d]; }
      m1[j] = s / HD; m2[j] = q / HD;
    }
    __syncthreads();
    float g[MBS];
    for (int i = 0; i < MBS; i++)
      g[i] = (gxh[i][j] - m1[i] - xh[i][j] * m2[i]) / sd[i];
    { // combined coefficients: tril(et * (attn + 1))
      const int i = j >> 3, k = j & 7;
      co[i][k] = (k <= i) ? tf[i] * lr8[k] * (1.f / HD) * (attn[i][k] + 1.f) : 0.f;
    }
    __syncthreads();
    float zb[MBS];
    for (int i = 0; i < MBS; i++) {          // Z1_bar column j
      float s = b1j;
      for (int d = 0; d < HD; d++) s += xq[i][d] * W1s[d][j];
#pragma unroll
      for (int k = 0; k < MBS; k++) s -= co[i][k] * g[k];
      zb[i] = s; zs[i][j] = s;
    }
    // fast-weight update (last_eta row)
    float sk[MBS];
    const float le = tf[MBS - 1] * (1.f / HD);
#pragma unroll
    for (int k = 0; k < MBS; k++) { sk[k] = le * lr8[k] * g[k]; b1j -= sk[k]; }
    for (int d = 0; d < HD; d++) {
      float a = 0.f;
#pragma unroll
      for (int k = 0; k < MBS; k++) a += sk[k] * xk[k][d];
      W1s[d][j] -= a;
    }
    __syncthreads();
    if (j < MBS) {                           // Z1_bar row stats
      float s = 0.f, q = 0.f;
      for (int d = 0; d < HD; d++) { const float v = zs[j][d]; s += v; q += v * v; }
      const float m = s / HD; mu[j] = m; sd[j] = sqrtf(q / HD - m * m + 1e-6f);
    }
    __syncthreads();
    for (int i = 0; i < MBS; i++) {
      const float o = xq[i][j] + (zb[i] - mu[i]) / sd[i] * gj + bj;
      outp[((size_t)(b * L_ + l0 + i)) * DM + h * HD + j] = o;
    }
    __syncthreads();
  }
}

// ======================================================================
// Window-softmax gain: one wave per token; lane-parallel dots + shfl_xor
// butterfly. gain = sigmoid(4 * max(softmax(scores)) - 2)
//   (max of softmax == 1 / sum(exp(s - max s)))
// ======================================================================
__global__ __launch_bounds__(256) void gain_kernel(const float* __restrict__ qg,
                                                   const float* __restrict__ kg,
                                                   float* __restrict__ gout) {
  const int wid = (int)((blockIdx.x * blockDim.x + threadIdx.x) >> 5);
  if (wid >= B_ * L_) return;
  const int lane = threadIdx.x & 31;
  const int b = wid / L_, l = wid % L_;
  const float* qr = qg + (size_t)wid * DM;
  float qv[16];
#pragma unroll
  for (int i = 0; i < 16; i++) qv[i] = qr[lane + 32 * i];
  float sc[2 * WINR + 1];
  float mx = -1e30f;
  const float scale = 1.f / sqrtf((float)DM);
#pragma unroll
  for (int w = 0; w < 2 * WINR + 1; w++) {
    int jj = l - WINR + w;
    jj = jj < 0 ? 0 : (jj > L_ - 1 ? L_ - 1 : jj);
    const float* kr = kg + ((size_t)(b * L_ + jj)) * DM;
    float s = 0.f;
#pragma unroll
    for (int i = 0; i < 16; i++) s += qv[i] * kr[lane + 32 * i];
#pragma unroll
    for (int o = 16; o > 0; o >>= 1) s += __shfl_xor(s, o, 32);
    s *= scale;
    sc[w] = s;
    mx = fmaxf(mx, s);
  }
  float se = 0.f;
#pragma unroll
  for (int w = 0; w < 2 * WINR + 1; w++) se += expf(sc[w] - mx);
  if (lane == 0) gout[wid] = sigm((1.f / se) * 4.f - 2.f);
}

// ======================================================================
// Final blend + residual copy
// ======================================================================
__global__ void blend_kernel(const float* __restrict__ zl, const float* __restrict__ zt,
                             const float* __restrict__ gain, const float* __restrict__ x,
                             float* __restrict__ out, float* __restrict__ resid) {
  int idx = blockIdx.x * blockDim.x + threadIdx.x;
  if (idx >= B_ * L_ * DM) return;
  const int row = idx / DM;
  const float g = gain[row];
  const float a = zl[idx];
  out[idx] = a + g * (zt[idx] - a);
  resid[idx] = x[idx];
}

// ======================================================================
extern "C" void kernel_launch(void* const* d_in, const int* in_sizes, int n_in,
                              void* d_out, int out_size, void* d_ws, size_t ws_size,
                              hipStream_t stream) {
  (void)in_sizes; (void)n_in; (void)out_size; (void)ws_size;
  const float* x        = (const float*)d_in[0];
  const float* ln_w     = (const float*)d_in[1];
  const float* ln_b     = (const float*)d_in[2];
  const float* in_proj  = (const float*)d_in[3];
  const float* conv_w   = (const float*)d_in[4];
  const float* conv_b   = (const float*)d_in[5];
  const float* x_proj   = (const float*)d_in[6];
  const float* dt_hw    = (const float*)d_in[7];
  const float* dt_hb    = (const float*)d_in[8];
  const float* D_param  = (const float*)d_in[9];
  const float* reduce_w = (const float*)d_in[10];
  const float* wq       = (const float*)d_in[11];
  const float* wk       = (const float*)d_in[12];
  const float* wv       = (const float*)d_in[13];
  const float* W1       = (const float*)d_in[14];
  const float* b1       = (const float*)d_in[15];
  const float* tnw      = (const float*)d_in[16];
  const float* tnb      = (const float*)d_in[17];
  const float* lr_w     = (const float*)d_in[18];
  const float* lr_b     = (const float*)d_in[19];
  const float* tdelta   = (const float*)d_in[20];
  const float* pn_w     = (const float*)d_in[21];
  const float* pn_b     = (const float*)d_in[22];
  const float* qnet_w   = (const float*)d_in[23];
  const float* gq_w     = (const float*)d_in[24];
  const float* gk_w     = (const float*)d_in[25];

  float* ws = (float*)d_ws;
  const size_t TOK = (size_t)B_ * L_;          // 8192
  const size_t NMO = TOK * DM;                 // 4194304
  // primary arenas
  float* xn   = ws;                             // TOK*DM
  float* xz   = xn   + NMO;                     // TOK*XZN
  float* xc   = xz   + TOK * XZN;               // TOK*DI
  float* xdbl = xc   + TOK * DI;                // TOK*XDN
  float* dtv  = xdbl + TOK * XDN;               // TOK*DI
  float* hred = dtv  + TOK * DI;                // TOK*DM
  // sub-arenas (reuse buffers that are dead after the SSM scan)
  float* ttt  = xz;                             // TOK*DM  (xz dead after scan)
  float* chid = xz + 1 * NMO;
  float* XQ   = xz + 2 * NMO;
  float* XK   = xz + 3 * NMO;
  float* XV   = xz + 4 * NMO;
  float* z_l  = dtv;                            // dtv dead after scan
  float* z_t  = dtv + 1 * NMO;
  float* qg   = dtv + 2 * NMO;
  float* kg   = dtv + 3 * NMO;
  float* lrv  = xdbl;                           // xdbl dead after scan (64K floats)

  float* out_p   = (float*)d_out;               // (B,L,DM)
  float* resid_p = out_p + NMO;                 // (B,L,DM)
  float* gain_p  = resid_p + NMO;               // (B,L,1)

  auto gemm = [&](const float* A, const float* Wm, float* C, int M, int N, int K, int lda) {
    dim3 g((N + BN - 1) / BN, (M + BM - 1) / BM);
    gemm_bf16_kernel<<<g, 256, 0, stream>>>(A, Wm, C, M, N, K, lda);
  };

  // 1) xn = LN(x)
  ln_kernel<<<(int)TOK, 256, 0, stream>>>(x, ln_w, ln_b, xn, DM);
  // 2) xz = xn @ in_proj^T
  gemm(xn, in_proj, xz, (int)TOK, XZN, DM, DM);
  // 3) depthwise conv + silu -> xc
  {
    int tot = B_ * L_ * DI;
    conv_silu_kernel<<<(tot + 255) / 256, 256, 0, stream>>>(xz, conv_w, conv_b, xc);
  }
  // 4) xdbl = xc @ x_proj^T   (N=48)
  gemm(xc, x_proj, xdbl, (int)TOK, XDN, DI, DI);
  // 5) dtv = dt_in @ dt_head^T  (A strided, lda=48, K=32)
  gemm(xdbl, dt_hw, dtv, (int)TOK, DI, DTR, XDN);
  // 6) Longhorn scan (fused silu(z) gating), y -> xc in place
  scan_kernel<<<dim3(DI / 256, B_), 256, 0, stream>>>(xc, dtv, xdbl, xz, dt_hb, D_param);
  // 7) h_red = y @ reduce^T
  gemm(xc, reduce_w, hred, (int)TOK, DM, DI, DI);
  // 8) XQ/XK/XV = xn @ w{q,k,v}^T
  gemm(xn, wq, XQ, (int)TOK, DM, DM, DM);
  gemm(xn, wk, XK, (int)TOK, DM, DM, DM);
  gemm(xn, wv, XV, (int)TOK, DM, DM, DM);
  // 9) rotary on XQ, XK
  {
    int tot = B_ * L_ * H_ * (HD / 2);
    rotary_kernel<<<(tot + 255) / 256, 256, 0, stream>>>(XQ);
    rotary_kernel<<<(tot + 255) / 256, 256, 0, stream>>>(XK);
  }
  // 10) lr values
  {
    int tot = B_ * L_ * H_;
    lr_kernel<<<(tot + 255) / 256, 256, 0, stream>>>(xn, lr_w, lr_b, lrv);
  }
  // 11) TTT-linear scan
  ttt_scan_kernel<<<B_ * H_, 64, 0, stream>>>(XQ, XK, XV, lrv, tdelta, W1, b1, tnw, tnb, ttt);
  // 12) post-norm
  ln_kernel<<<(int)TOK, 256, 0, stream>>>(ttt, pn_w, pn_b, chid, DM);
  // 13) output projections
  gemm(hred, qnet_w, z_l, (int)TOK, DM, DM, DM);
  gemm(chid, qnet_w, z_t, (int)TOK, DM, DM, DM);
  gemm(xn, gq_w, qg, (int)TOK, DM, DM, DM);
  gemm(xn, gk_w, kg, (int)TOK, DM, DM, DM);
  // 14) windowed-softmax gain -> gain output region (wave per token)
  {
    int waves = (int)TOK;
    int threads = waves * 32;
    gain_kernel<<<(threads + 255) / 256, 256, 0, stream>>>(qg, kg, gain_p);
  }
  // 15) blend + residual copy
  {
    int tot = B_ * L_ * DM;
    blend_kernel<<<(tot + 255) / 256, 256, 0, stream>>>(z_l, z_t, gain_p, x, out_p, resid_p);
  }
}
